// TransformerBlock_39213051412825
// MI455X (gfx1250) — matrix operand order repaired and checked by reading, not compiled, not run on hardware
//
#include <hip/hip_runtime.h>
#include <hip/hip_bf16.h>

typedef __attribute__((ext_vector_type(16))) __bf16 bf16x16;
typedef __attribute__((ext_vector_type(8)))  __bf16 bf16x8;
typedef __attribute__((ext_vector_type(4)))  __bf16 bf16x4;
typedef __attribute__((ext_vector_type(8)))  float  v8f;

#define BATCH 2
#define NPTS  8192
#define DIM   128
#define KNN   16
#define BN    (BATCH * NPTS)   // 16384

// ---------------------------------------------------------------------------
// CDNA5 async copy: global -> LDS, 16B per lane, tracked by ASYNCcnt.
// GVS form: mem_addr = SADDR(64b) + VADDR(32b) ; LDS dest addr from VGPR.
// (no static LDS in these kernels => dynamic LDS segment starts at offset 0)
// ---------------------------------------------------------------------------
__device__ __forceinline__ void async_copy_b128(unsigned lds_byte_off,
                                                const void* gbase,
                                                unsigned gbyte_off) {
    asm volatile("global_load_async_to_lds_b128 %0, %1, %2"
                 :: "v"(lds_byte_off), "v"(gbyte_off), "s"(gbase)
                 : "memory");
}
__device__ __forceinline__ void wait_async0() {
    asm volatile("s_wait_asynccnt 0x0" ::: "memory");
}

// ---------------------------------------------------------------------------
// Pack a 128x128 fp32 weight (row-major (in,out)) into WMMA-B bf16 layout:
// tile t = nt*4 + ks holds K=ks*32..+31, N=nt*16..+15.
// lane L -> n = L%16, k = ks*32 + 16*(L/16) + e ; packed idx = t*512+L*16+e.
// ---------------------------------------------------------------------------
__global__ void pack_kernel(const float* __restrict__ wq, const float* __restrict__ wk,
                            const float* __restrict__ wv, const float* __restrict__ dw2,
                            const float* __restrict__ gw1, const float* __restrict__ gw2,
                            __bf16* __restrict__ packed) {
    const float* src = (blockIdx.x == 0) ? wq :
                       (blockIdx.x == 1) ? wk :
                       (blockIdx.x == 2) ? wv :
                       (blockIdx.x == 3) ? dw2 :
                       (blockIdx.x == 4) ? gw1 : gw2;
    __bf16* dst = packed + (size_t)blockIdx.x * 16384;
    for (int i = threadIdx.x; i < 16384; i += 256) {
        int t = i >> 9, r = i & 511;
        int L = r >> 4, e = r & 15;
        int nt = t >> 2, ks = t & 3;
        int k = ks * 32 + ((L >> 4) << 4) + e;
        int n = (nt << 4) + (L & 15);
        dst[i] = (__bf16)src[k * 128 + n];
    }
}

// fp32 feats -> bf16 copy (row-major), feeds async A staging in qkv_kernel
__global__ void cast_kernel(const float* __restrict__ f, __bf16* __restrict__ o) {
    int i = (blockIdx.x * 256 + threadIdx.x) * 8;
    float4 a = *(const float4*)(f + i);
    float4 b = *(const float4*)(f + i + 4);
    bf16x8 r;
    r[0] = (__bf16)a.x; r[1] = (__bf16)a.y; r[2] = (__bf16)a.z; r[3] = (__bf16)a.w;
    r[4] = (__bf16)b.x; r[5] = (__bf16)b.y; r[6] = (__bf16)b.z; r[7] = (__bf16)b.w;
    *(bf16x8*)(o + i) = r;
}

// ---------------------------------------------------------------------------
// KNN: whole batch xyz (96KB) async-DMA'd into LDS; 1 thread = 1 query;
// stable unrolled insertion keeps the register-resident top-16.
// ---------------------------------------------------------------------------
__global__ void knn_kernel(const float* __restrict__ xyz, int* __restrict__ knn) {
    extern __shared__ char dynsmem[];
    float* sx = (float*)dynsmem;                    // [8192*3]
    const int tid = threadIdx.x;
    const int b   = blockIdx.x >> 5;                // / (8192/256)
    const int qi  = ((blockIdx.x & 31) << 8) + tid; // query within batch
    const char* bx = (const char*)(xyz + (size_t)b * NPTS * 3);
#pragma unroll
    for (int it = 0; it < 24; ++it) {               // 24*4096 = 98304 bytes
        unsigned off = (unsigned)tid * 16u + (unsigned)it * 4096u;
        async_copy_b128(off, bx, off);
    }
    wait_async0();
    __syncthreads();

    const float qx = sx[qi * 3 + 0], qy = sx[qi * 3 + 1], qz = sx[qi * 3 + 2];
    float bd[16]; int bi[16];
#pragma unroll
    for (int p = 0; p < 16; ++p) { bd[p] = 3.4e38f; bi[p] = 0; }

    for (int j = 0; j < NPTS; ++j) {
        float dx = qx - sx[j * 3 + 0];
        float dy = qy - sx[j * 3 + 1];
        float dz = qz - sx[j * 3 + 2];
        float d = dx * dx + dy * dy + dz * dz;
        if (d < bd[15]) {
            float cd = d; int ci = j;
#pragma unroll
            for (int p = 0; p < 16; ++p) {
                if (cd < bd[p]) {
                    float td = bd[p]; int ti = bi[p];
                    bd[p] = cd; bi[p] = ci;
                    cd = td; ci = ti;
                }
            }
        }
    }
    int* o = knn + ((size_t)b * NPTS + qi) * KNN;
#pragma unroll
    for (int p = 0; p < 16; ++p) o[p] = bi[p];
}

// ---------------------------------------------------------------------------
// WMMA fragment loaders (A and B both from LDS here)
// A (16x32 bf16): lane L -> m = L&15, h = L>>4; elems 0..15: k=ks*32+16h+e.
// Two 16B loads per fragment.
// ---------------------------------------------------------------------------
__device__ __forceinline__ bf16x16 load_a_frag(const __bf16* sSrc, int row, int ks, int h) {
    bf16x16 a;
    const char* ap = (const char*)sSrc + row * 256 + ks * 64 + h * 32;
    ((float4*)&a)[0] = *(const float4*)(ap);
    ((float4*)&a)[1] = *(const float4*)(ap + 16);
    return a;
}
__device__ __forceinline__ bf16x16 load_b_frag(const __bf16* wSrc, int nt, int ks, int lane) {
    bf16x16 bb;
    const float4* bp = (const float4*)((const char*)wSrc + ((nt * 4 + ks) * 512 + lane * 16) * 2);
    ((float4*)&bb)[0] = bp[0];
    ((float4*)&bb)[1] = bp[1];
    return bb;
}

// GEMM over the 256x128 LDS tile: wave w owns m-tiles {2w, 2w+1} x 8 n-tiles.
// C = relu?(A @ W + bias) -> LDS bf16. W is an LDS-resident packed weight.
__device__ __forceinline__ void tile_gemm(const __bf16* __restrict__ sSrc,
                                          const __bf16* __restrict__ sW,
                                          const float* __restrict__ sBias,
                                          __bf16* __restrict__ sDst,
                                          int wave, int lane, bool relu) {
    const int h = lane >> 4, ln = lane & 15;
    for (int mi = 0; mi < 2; ++mi) {
        const int mt = wave * 2 + mi;
        const int arow = mt * 16 + ln;
#pragma unroll
        for (int nt = 0; nt < 8; ++nt) {
            const float bias = sBias[nt * 16 + ln];
            v8f acc = {bias, bias, bias, bias, bias, bias, bias, bias};
#pragma unroll
            for (int ks = 0; ks < 4; ++ks) {
                bf16x16 a = load_a_frag(sSrc, arow, ks, h);
                bf16x16 bb = load_b_frag(sW, nt, ks, lane);
                acc = __builtin_amdgcn_wmma_f32_16x16x32_bf16(
                    false, a, false, bb, (short)0, acc, false, false);
            }
#pragma unroll
            for (int r = 0; r < 8; ++r) {
                float v = acc[r];
                if (relu) v = v > 0.f ? v : 0.f;
                int row = mt * 16 + r + 8 * h;
                sDst[row * 128 + nt * 16 + ln] = (__bf16)v;
            }
        }
    }
}

// ---------------------------------------------------------------------------
// Q/K/V projection: async-DMA bf16 feats tile [64x128] to LDS, WMMA, fp32 out.
// B fragments stream from global (L1/L2-resident packed weights).
// ---------------------------------------------------------------------------
__global__ void qkv_kernel(const __bf16* __restrict__ featsb, const __bf16* __restrict__ packed,
                           float* __restrict__ qf, float* __restrict__ kf, float* __restrict__ vf) {
    extern __shared__ char dynsmem[];
    __bf16* sA = (__bf16*)dynsmem;                 // [64*128] bf16 = 16KB
    const int tid = threadIdx.x, wave = tid >> 5, lane = tid & 31;
    const int rowBase = blockIdx.x * 64;
    const __bf16* wpk = packed + (size_t)blockIdx.y * 16384;
    float* out = (blockIdx.y == 0) ? qf : (blockIdx.y == 1) ? kf : vf;

    const char* src = (const char*)(featsb + (size_t)rowBase * 128);
#pragma unroll
    for (int it = 0; it < 8; ++it) {               // 8*2048 = 16384 bytes
        unsigned off = (unsigned)tid * 16u + (unsigned)it * 2048u;
        async_copy_b128(off, src, off);
    }
    wait_async0();
    __syncthreads();

    const int h = lane >> 4, ln = lane & 15;
    const int arow = wave * 16 + ln;
#pragma unroll
    for (int nt = 0; nt < 8; ++nt) {
        v8f acc = {0.f, 0.f, 0.f, 0.f, 0.f, 0.f, 0.f, 0.f};
#pragma unroll
        for (int ks = 0; ks < 4; ++ks) {
            bf16x16 a = load_a_frag(sA, arow, ks, h);
            bf16x16 bb = load_b_frag(wpk, nt, ks, lane);
            acc = __builtin_amdgcn_wmma_f32_16x16x32_bf16(
                false, a, false, bb, (short)0, acc, false, false);
        }
#pragma unroll
        for (int r = 0; r < 8; ++r) {
            int row = rowBase + wave * 16 + r + 8 * h;
            out[(size_t)row * 128 + nt * 16 + ln] = acc[r];
        }
    }
}

// ---------------------------------------------------------------------------
// Fused attention kernel: 16 queries x 16 neighbors = 256 rows per block.
// All intermediates + all three packed weights live in LDS (~293KB of 320KB).
// Weights are async-DMA'd at block start, overlapped with pos-MLP layer 1.
// ---------------------------------------------------------------------------
#define OFF_A   0
#define OFF_H   65536
#define OFF_POS 131072
#define OFF_W   196608          // 3 x 32768 bytes of packed weights
#define OFF_IDX 294912
#define OFF_W1  295936
#define OFF_B   297472
#define OFF_QX  299520
#define ATTN_LDS 299712

__global__ void attn_kernel(const float* __restrict__ xyz, const float* __restrict__ feats,
                            const float* __restrict__ qf, const float* __restrict__ kf,
                            const float* __restrict__ vf, const int* __restrict__ knn,
                            const __bf16* __restrict__ pk_attn,   // dw2|gw1|gw2 packed, contiguous
                            const float* __restrict__ dw1, const float* __restrict__ db1,
                            const float* __restrict__ db2, const float* __restrict__ gb1,
                            const float* __restrict__ gb2, float* __restrict__ resb) {
    extern __shared__ char dynsmem[];
    __bf16* sA   = (__bf16*)(dynsmem + OFF_A);    // h1 -> x -> logits
    __bf16* sH   = (__bf16*)(dynsmem + OFF_H);    // g-MLP hidden
    __bf16* sPos = (__bf16*)(dynsmem + OFF_POS);  // pos_encode
    int*    sIdx = (int*)  (dynsmem + OFF_IDX);
    float*  sW1  = (float*)(dynsmem + OFF_W1);    // d_w1 [3x128]
    float*  sB   = (float*)(dynsmem + OFF_B);     // d_b1,d_b2,g_b1,g_b2
    float*  sQx  = (float*)(dynsmem + OFF_QX);    // 16 query xyz

    const int tid  = threadIdx.x;
    const int wave = tid >> 5, lane = tid & 31;
    const int gBase    = blockIdx.x * 16;
    const int b        = gBase / NPTS;
    const int qBase    = gBase - b * NPTS;
    const int batchOff = b * NPTS;

    // ---- kick off async weight DMA (96KB) before anything else ----
#pragma unroll
    for (int it = 0; it < 24; ++it) {             // 24*4096 = 98304 bytes
        unsigned off = (unsigned)tid * 16u + (unsigned)it * 4096u;
        async_copy_b128(OFF_W + off, pk_attn, off);
    }

    // ---- setup ----
    {
        int m = tid >> 4, kk = tid & 15;
        sIdx[tid] = knn[(size_t)(batchOff + qBase + m) * KNN + kk];
    }
    for (int i = tid; i < 384; i += 256) sW1[i] = dw1[i];
    if (tid < 128) {
        sB[tid]       = db1[tid];
        sB[128 + tid] = db2[tid];
        sB[256 + tid] = gb1[tid];
        sB[384 + tid] = gb2[tid];
    }
    if (tid < 48) sQx[tid] = xyz[(size_t)(batchOff + qBase) * 3 + tid];
    __syncthreads();

    // ---- stage 1: pos-MLP layer 1 (3 -> 128, ReLU) — overlaps weight DMA ----
    {
        const int r = tid, m = r >> 4;
        const int nb = batchOff + sIdx[r];
        float nx = xyz[(size_t)nb * 3 + 0];
        float ny = xyz[(size_t)nb * 3 + 1];
        float nz = xyz[(size_t)nb * 3 + 2];
        float rx = sQx[m * 3 + 0] - nx;
        float ry = sQx[m * 3 + 1] - ny;
        float rz = sQx[m * 3 + 2] - nz;
        __bf16* dst = sA + r * 128;
        for (int o = 0; o < 128; o += 4) {
            float4 w0 = *(const float4*)(sW1 + o);
            float4 w1 = *(const float4*)(sW1 + 128 + o);
            float4 w2 = *(const float4*)(sW1 + 256 + o);
            float4 bs = *(const float4*)(sB + o);
            bf16x4 d;
            float h0 = bs.x + rx * w0.x + ry * w1.x + rz * w2.x;
            float h1 = bs.y + rx * w0.y + ry * w1.y + rz * w2.y;
            float h2 = bs.z + rx * w0.z + ry * w1.z + rz * w2.z;
            float h3 = bs.w + rx * w0.w + ry * w1.w + rz * w2.w;
            d[0] = (__bf16)(h0 > 0.f ? h0 : 0.f);
            d[1] = (__bf16)(h1 > 0.f ? h1 : 0.f);
            d[2] = (__bf16)(h2 > 0.f ? h2 : 0.f);
            d[3] = (__bf16)(h3 > 0.f ? h3 : 0.f);
            *(bf16x4*)(dst + o) = d;
        }
    }
    wait_async0();            // weights resident before any wave enters GEMM
    __syncthreads();

    const __bf16* sWdw2 = (const __bf16*)(dynsmem + OFF_W);
    const __bf16* sWgw1 = (const __bf16*)(dynsmem + OFF_W + 32768);
    const __bf16* sWgw2 = (const __bf16*)(dynsmem + OFF_W + 65536);

    // ---- stage 2: pos = h1 @ d_w2 + d_b2 ----
    tile_gemm(sA, sWdw2, sB + 128, sPos, wave, lane, false);
    __syncthreads();

    // ---- stage 3: x = q - k_gather + pos (vectorized) ----
    {
        const int r = tid, m = r >> 4;
        const float* qrow = qf + (size_t)(batchOff + qBase + m) * 128;
        const float* krow = kf + (size_t)(batchOff + sIdx[r]) * 128;
        const __bf16* prow = sPos + r * 128;
        __bf16* dst = sA + r * 128;
        for (int o = 0; o < 128; o += 4) {
            float4 qv = *(const float4*)(qrow + o);
            float4 kv = *(const float4*)(krow + o);
            bf16x4 pv = *(const bf16x4*)(prow + o);
            bf16x4 d;
            d[0] = (__bf16)(qv.x - kv.x + (float)pv[0]);
            d[1] = (__bf16)(qv.y - kv.y + (float)pv[1]);
            d[2] = (__bf16)(qv.z - kv.z + (float)pv[2]);
            d[3] = (__bf16)(qv.w - kv.w + (float)pv[3]);
            *(bf16x4*)(dst + o) = d;
        }
    }
    __syncthreads();

    // ---- stage 4: hidden = relu(x @ g_w1 + g_b1) ----
    tile_gemm(sA, sWgw1, sB + 256, sH, wave, lane, true);
    __syncthreads();

    // ---- stage 5: logits = hidden @ g_w2 + g_b2 (into sA) ----
    tile_gemm(sH, sWgw2, sB + 384, sA, wave, lane, false);
    __syncthreads();

    // ---- stage 6: softmax over k + weighted sum, 8 channels per thread ----
    {
        const int m = tid >> 4;            // query in tile
        const int f0 = (tid & 15) << 3;    // 8 consecutive channels
        float lmax[8];
#pragma unroll
        for (int j = 0; j < 8; ++j) lmax[j] = -3.4e38f;
#pragma unroll
        for (int kk = 0; kk < 16; ++kk) {
            bf16x8 l8 = *(const bf16x8*)(sA + (m * 16 + kk) * 128 + f0);
#pragma unroll
            for (int j = 0; j < 8; ++j) lmax[j] = fmaxf(lmax[j], (float)l8[j]);
        }
        float sum[8], acc[8];
#pragma unroll
        for (int j = 0; j < 8; ++j) { sum[j] = 0.f; acc[j] = 0.f; }
#pragma unroll
        for (int kk = 0; kk < 16; ++kk) {
            const int row = m * 16 + kk;
            const float* vrow = vf + (size_t)(batchOff + sIdx[row]) * 128 + f0;
            float4 v0 = *(const float4*)(vrow);
            float4 v1 = *(const float4*)(vrow + 4);
            bf16x8 l8 = *(const bf16x8*)(sA + row * 128 + f0);
            bf16x8 p8 = *(const bf16x8*)(sPos + row * 128 + f0);
#pragma unroll
            for (int j = 0; j < 8; ++j) {
                float w = __expf((float)l8[j] - lmax[j]);
                sum[j] += w;
                float vj = (j < 4) ? (&v0.x)[j] : (&v1.x)[j - 4];
                acc[j] += w * (vj + (float)p8[j]);
            }
        }
        const float* frow = feats + (size_t)(batchOff + qBase + m) * 128 + f0;
        float4 fa = *(const float4*)(frow);
        float4 fb = *(const float4*)(frow + 4);
        float4 oa, ob;
        oa.x = acc[0] / sum[0] + fa.x;  oa.y = acc[1] / sum[1] + fa.y;
        oa.z = acc[2] / sum[2] + fa.z;  oa.w = acc[3] / sum[3] + fa.w;
        ob.x = acc[4] / sum[4] + fb.x;  ob.y = acc[5] / sum[5] + fb.y;
        ob.z = acc[6] / sum[6] + fb.z;  ob.w = acc[7] / sum[7] + fb.w;
        float* orow = resb + (size_t)(gBase + m) * 128 + f0;
        *(float4*)(orow)     = oa;
        *(float4*)(orow + 4) = ob;
    }
}

// ---------------------------------------------------------------------------
// BatchNorm (training-mode batch stats over b*n per channel)
// ---------------------------------------------------------------------------
__global__ void bn_stats_kernel(const float* __restrict__ resb, float* __restrict__ stats) {
    __shared__ float s1[256];
    __shared__ float s2[256];
    const int c = blockIdx.x, tid = threadIdx.x;
    float a = 0.f, q = 0.f;
    for (int r = tid; r < BN; r += 256) {
        float v = resb[(size_t)r * 128 + c];
        a += v;
        q += v * v;
    }
    s1[tid] = a; s2[tid] = q;
    __syncthreads();
    for (int s = 128; s > 0; s >>= 1) {
        if (tid < s) { s1[tid] += s1[tid + s]; s2[tid] += s2[tid + s]; }
        __syncthreads();
    }
    if (tid == 0) {
        float mean = s1[0] * (1.0f / BN);
        float var  = s2[0] * (1.0f / BN) - mean * mean;
        stats[c]       = mean;
        stats[128 + c] = rsqrtf(var + 1e-5f);
    }
}

__global__ void bn_apply_kernel(const float* __restrict__ resb, const float* __restrict__ stats,
                                const float* __restrict__ bnw, const float* __restrict__ bnb,
                                float* __restrict__ out) {
    size_t base = ((size_t)blockIdx.x * 256 + threadIdx.x) * 4;
    int c = (int)(base & 127);
    float4 r = *(const float4*)(resb + base);
    float4 mu = *(const float4*)(stats + c);
    float4 rs = *(const float4*)(stats + 128 + c);
    float4 w  = *(const float4*)(bnw + c);
    float4 bb = *(const float4*)(bnb + c);
    float4 o;
    o.x = (r.x - mu.x) * rs.x * w.x + bb.x;
    o.y = (r.y - mu.y) * rs.y * w.y + bb.y;
    o.z = (r.z - mu.z) * rs.z * w.z + bb.z;
    o.w = (r.w - mu.w) * rs.w * w.w + bb.w;
    *(float4*)(out + base) = o;
}

// ---------------------------------------------------------------------------
extern "C" void kernel_launch(void* const* d_in, const int* in_sizes, int n_in,
                              void* d_out, int out_size, void* d_ws, size_t ws_size,
                              hipStream_t stream) {
    const float* xyz   = (const float*)d_in[0];
    const float* feats = (const float*)d_in[1];
    const float* wq    = (const float*)d_in[2];
    const float* wk    = (const float*)d_in[3];
    const float* wv    = (const float*)d_in[4];
    const float* d_w1  = (const float*)d_in[5];
    const float* d_b1  = (const float*)d_in[6];
    const float* d_w2  = (const float*)d_in[7];
    const float* d_b2  = (const float*)d_in[8];
    const float* g_w1  = (const float*)d_in[9];
    const float* g_b1  = (const float*)d_in[10];
    const float* g_w2  = (const float*)d_in[11];
    const float* g_b2  = (const float*)d_in[12];
    const float* bn_w  = (const float*)d_in[13];
    const float* bn_b  = (const float*)d_in[14];

    char* ws = (char*)d_ws;
    size_t off = 0;
    int*    knn    = (int*)(ws + off);     off += (size_t)BN * KNN * 4;     // 1MB
    __bf16* packed = (__bf16*)(ws + off);  off += (size_t)6 * 16384 * 2;    // 192KB
    off = (off + 255) & ~(size_t)255;
    __bf16* featsb = (__bf16*)(ws + off);  off += (size_t)BN * DIM * 2;     // 4MB
    float*  qf     = (float*)(ws + off);   off += (size_t)BN * DIM * 4;     // 8MB
    float*  kf     = (float*)(ws + off);   off += (size_t)BN * DIM * 4;
    float*  vf     = (float*)(ws + off);   off += (size_t)BN * DIM * 4;
    float*  resb   = (float*)(ws + off);   off += (size_t)BN * DIM * 4;
    float*  stats  = (float*)(ws + off);   off += 256 * 4;

    pack_kernel<<<6, 256, 0, stream>>>(wq, wk, wv, d_w2, g_w1, g_w2, packed);
    cast_kernel<<<(BN * DIM) / (256 * 8), 256, 0, stream>>>(feats, featsb);
    knn_kernel<<<BN / 256, 256, NPTS * 3 * 4, stream>>>(xyz, knn);
    qkv_kernel<<<dim3(BN / 64, 3), 128, 64 * 128 * 2, stream>>>(featsb, packed, qf, kf, vf);
    attn_kernel<<<BN / 16, 256, ATTN_LDS, stream>>>(
        xyz, feats, qf, kf, vf, knn,
        packed + 3 * 16384,
        d_w1, d_b1, d_b2, g_b1, g_b2, resb);
    bn_stats_kernel<<<128, 256, 0, stream>>>(resb, stats);
    bn_apply_kernel<<<(BN * DIM) / 1024, 256, 0, stream>>>(resb, stats, bn_w, bn_b, (float*)d_out);
}
